// MambaBlock_48679159333165
// MI455X (gfx1250) — compile-verified, matrix operand order checked by reading, not run on hardware
//
#include <hip/hip_runtime.h>
#include <hip/hip_bf16.h>

// ---------------------------------------------------------------------------
// MambaBlock on gfx1250 (MI455X):
//   out = x + scan(dA,dBx,C) + xn*D_skip
// Heavy GEMM (xn @ W_delta, 8.6 GFLOP) via v_wmma_f32_16x16x32_bf16 with a
// 32x32 per-wave tile: 2 A-frags x 2 B-frags -> 4 WMMAs per K-step, halving
// L2 fragment traffic vs 16x16 tiles (8 -> 16 FLOP/byte) and giving 4
// independent accumulator chains for latency hiding.
// Working set (~45 MB) fits in 192 MB L2 -> stream fragments from L2.
// Scan parallelized one lane per (b,d,n): 1024 wave32s, shfl-xor reduction.
// ---------------------------------------------------------------------------

typedef __attribute__((ext_vector_type(16))) __bf16       v16bf;
typedef __attribute__((ext_vector_type(8)))  float        v8f;
typedef __attribute__((ext_vector_type(4)))  unsigned int v4u;
typedef __attribute__((ext_vector_type(8)))  unsigned int v8u;

#define BSZ   2
#define LSEQ  2048
#define DDIM  1024
#define NST   16
#define MROWS (BSZ * LSEQ)       // 4096
#define NTOT  (DDIM + 2 * NST)   // 1056 (delta cols | B cols | C cols)

__device__ __forceinline__ unsigned short f32_to_bf16(float f) {
    unsigned int u = __float_as_uint(f);
    unsigned int rnd = 0x7FFFu + ((u >> 16) & 1u);   // round-to-nearest-even
    return (unsigned short)((u + rnd) >> 16);
}

// ---------------------------------------------------------------------------
// Kernel 0: pack W_delta (D x D), W_B (D x 16), W_C (D x 16) into bf16
// Wt[NTOT][DDIM], row n = output column n (so B fragments load row-major).
// ---------------------------------------------------------------------------
__global__ void __launch_bounds__(256)
pack_w_kernel(const float* __restrict__ Wd, const float* __restrict__ WB,
              const float* __restrict__ WC, unsigned short* __restrict__ Wt) {
    int idx = blockIdx.x * 256 + threadIdx.x;           // over NTOT*DDIM
    if (idx >= NTOT * DDIM) return;
    int n = idx >> 10;          // 0..1055
    int k = idx & (DDIM - 1);   // 0..1023
    float v;
    if (n < DDIM)            v = Wd[(size_t)k * DDIM + n];
    else if (n < DDIM + NST) v = WB[(size_t)k * NST + (n - DDIM)];
    else                     v = WC[(size_t)k * NST + (n - DDIM - NST)];
    Wt[(size_t)n * DDIM + k] = f32_to_bf16(v);
}

// ---------------------------------------------------------------------------
// Kernel 1: LayerNorm; one block per row. Writes f32 (scan) + bf16 (WMMA).
// ---------------------------------------------------------------------------
__global__ void __launch_bounds__(256)
ln_kernel(const float* __restrict__ x, const float* __restrict__ gamma,
          const float* __restrict__ beta, float* __restrict__ xn,
          unsigned short* __restrict__ xnb) {
    __shared__ float rs[256], rs2[256];
    const int row = blockIdx.x, tid = threadIdx.x;
    const float* xr = x + (size_t)row * DDIM;
    float s = 0.f, s2 = 0.f;
#pragma unroll
    for (int i = 0; i < DDIM / 256; ++i) {
        float v = xr[tid + i * 256];
        s += v; s2 += v * v;
    }
    rs[tid] = s; rs2[tid] = s2;
    __syncthreads();
    for (int off = 128; off > 0; off >>= 1) {
        if (tid < off) { rs[tid] += rs[tid + off]; rs2[tid] += rs2[tid + off]; }
        __syncthreads();
    }
    const float mu   = rs[0] * (1.f / DDIM);
    const float var  = rs2[0] * (1.f / DDIM) - mu * mu;
    const float rstd = rsqrtf(var + 1e-5f);
#pragma unroll
    for (int i = 0; i < DDIM / 256; ++i) {
        int c = tid + i * 256;
        float v = (xr[c] - mu) * rstd * gamma[c] + beta[c];
        xn [(size_t)row * DDIM + c] = v;
        xnb[(size_t)row * DDIM + c] = f32_to_bf16(v);
    }
}

// ---------------------------------------------------------------------------
// Kernel 2: WMMA GEMM  C[M x NTOT] = xn_bf16[M x K] * Wt^T   (K = 1024)
// One wave per 32x32 tile (4 accumulators). Fragment layout per CDNA5 ISA
// (16-bit A 16x32): lane&15 = row, lane>>4 selects K-half; two contiguous
// 16-byte chunks at kbase and kbase+16 elements.
// ---------------------------------------------------------------------------
__device__ __forceinline__ v16bf load_frag(const unsigned short* p) {
    v4u lo = *reinterpret_cast<const v4u*>(p);        // K = base .. base+7
    v4u hi = *reinterpret_cast<const v4u*>(p + 16);   // K = base+16 .. base+23
    v8u t;
    t[0] = lo[0]; t[1] = lo[1]; t[2] = lo[2]; t[3] = lo[3];
    t[4] = hi[0]; t[5] = hi[1]; t[6] = hi[2]; t[7] = hi[3];
    return __builtin_bit_cast(v16bf, t);
}

__device__ __forceinline__ float softplus_f(float v) {
    return (v > 20.f) ? v : log1pf(__expf(v));
}

// Epilogue for one 16x16 sub-tile; routing is wave-uniform (n0 16-aligned).
__device__ __forceinline__ void
epilogue_tile(v8f c, int m0, int n0, int lane,
              const float* __restrict__ b_delta, const float* __restrict__ b_B,
              const float* __restrict__ b_C, float* __restrict__ delta_out,
              float* __restrict__ bc_out) {
    const int lrow  = lane & 15;
    const int n     = n0 + lrow;
    const int mbase = m0 + ((lane >> 4) ? 8 : 0);
    if (n0 < DDIM) {                         // delta path
        const float bias = b_delta[n];
#pragma unroll
        for (int r = 0; r < 8; ++r)
            delta_out[(size_t)(mbase + r) * DDIM + n] = softplus_f(c[r] + bias);
    } else if (n0 < DDIM + NST) {            // Bm path
        const float bias = b_B[n - DDIM];
#pragma unroll
        for (int r = 0; r < 8; ++r)
            bc_out[(size_t)(mbase + r) * 32 + (n - DDIM)] = c[r] + bias;
    } else {                                 // Cm path
        const float bias = b_C[n - DDIM - NST];
#pragma unroll
        for (int r = 0; r < 8; ++r)
            bc_out[(size_t)(mbase + r) * 32 + 16 + (n - DDIM - NST)] = c[r] + bias;
    }
}

__global__ void __launch_bounds__(128)
gemm_kernel(const unsigned short* __restrict__ A,   // [MROWS][DDIM] bf16
            const unsigned short* __restrict__ Wt,  // [NTOT][DDIM] bf16
            const float* __restrict__ b_delta, const float* __restrict__ b_B,
            const float* __restrict__ b_C,
            float* __restrict__ delta_out,          // [MROWS][DDIM]
            float* __restrict__ bc_out) {           // [MROWS][32]  (B|C)
    const int wave = threadIdx.x >> 5;
    const int lane = threadIdx.x & 31;
    const int tile = blockIdx.x * 4 + wave;         // grid sized exactly
    const int tn2 = tile % (NTOT / 32);             // 0..32
    const int tm2 = tile / (NTOT / 32);             // 0..127
    const int m0 = tm2 * 32, n0 = tn2 * 32;
    const int lrow  = lane & 15;
    const int kskew = (lane >> 4) ? 8 : 0;

    const unsigned short* arow0 = A  + (size_t)(m0 + lrow)      * DDIM + kskew;
    const unsigned short* arow1 = A  + (size_t)(m0 + 16 + lrow) * DDIM + kskew;
    const unsigned short* brow0 = Wt + (size_t)(n0 + lrow)      * DDIM + kskew;
    const unsigned short* brow1 = Wt + (size_t)(n0 + 16 + lrow) * DDIM + kskew;

    v8f c00 = {}; v8f c01 = {}; v8f c10 = {}; v8f c11 = {};
    for (int k = 0; k < DDIM; k += 32) {
        v16bf a0 = load_frag(arow0 + k);
        v16bf a1 = load_frag(arow1 + k);
        v16bf b0 = load_frag(brow0 + k);
        v16bf b1 = load_frag(brow1 + k);
        c00 = __builtin_amdgcn_wmma_f32_16x16x32_bf16(false, a0, false, b0,
                                                      (short)0, c00, false, false);
        c01 = __builtin_amdgcn_wmma_f32_16x16x32_bf16(false, a0, false, b1,
                                                      (short)0, c01, false, false);
        c10 = __builtin_amdgcn_wmma_f32_16x16x32_bf16(false, a1, false, b0,
                                                      (short)0, c10, false, false);
        c11 = __builtin_amdgcn_wmma_f32_16x16x32_bf16(false, a1, false, b1,
                                                      (short)0, c11, false, false);
    }

    epilogue_tile(c00, m0,      n0,      lane, b_delta, b_B, b_C, delta_out, bc_out);
    epilogue_tile(c01, m0,      n0 + 16, lane, b_delta, b_B, b_C, delta_out, bc_out);
    epilogue_tile(c10, m0 + 16, n0,      lane, b_delta, b_B, b_C, delta_out, bc_out);
    epilogue_tile(c11, m0 + 16, n0 + 16, lane, b_delta, b_B, b_C, delta_out, bc_out);
}

// ---------------------------------------------------------------------------
// Kernel 3: selective scan. Lane = (b, d, n). 32768 threads = 1024 wave32s.
// h_{l} = exp(delta*A)*h_{l-1} + delta*B_n*xn ; y = sum_n h*C_n (shfl-xor).
// Next-step loads software-pipelined ahead of the dependent update.
// ---------------------------------------------------------------------------
__global__ void __launch_bounds__(256)
scan_kernel(const float* __restrict__ x, const float* __restrict__ xn,
            const float* __restrict__ delta, const float* __restrict__ bc,
            const float* __restrict__ A_log, const float* __restrict__ D_skip,
            float* __restrict__ out) {
    const int gid = blockIdx.x * 256 + threadIdx.x;
    const int n = gid & (NST - 1);
    const int d = (gid >> 4) & (DDIM - 1);
    const int b = gid >> 14;

    const float Ac  = -__expf(A_log[d * NST + n]);
    const float dsk = D_skip[d];
    const size_t baseD = (size_t)b * LSEQ * DDIM + d;
    const size_t baseN = (size_t)b * LSEQ * 32 + n;

    float h = 0.f;
    float dlt = delta[baseD];
    float xv  = xn[baseD];
    float Bn  = bc[baseN];
    float Cn  = bc[baseN + 16];

    for (int l = 0; l < LSEQ; ++l) {
        float dlt_n = 0.f, xv_n = 0.f, Bn_n = 0.f, Cn_n = 0.f;
        if (l + 1 < LSEQ) {                  // prefetch next step
            size_t iD = baseD + (size_t)(l + 1) * DDIM;
            size_t iN = baseN + (size_t)(l + 1) * 32;
            dlt_n = delta[iD]; xv_n = xn[iD];
            Bn_n  = bc[iN];    Cn_n = bc[iN + 16];
        }
        float dA = __expf(dlt * Ac);
        h = fmaf(dA, h, dlt * Bn * xv);
        float p = h * Cn;
        p += __shfl_xor(p, 8, 16);
        p += __shfl_xor(p, 4, 16);
        p += __shfl_xor(p, 2, 16);
        p += __shfl_xor(p, 1, 16);
        if (n == 0) {
            size_t iD = baseD + (size_t)l * DDIM;
            out[iD] = x[iD] + p + xv * dsk;
        }
        dlt = dlt_n; xv = xv_n; Bn = Bn_n; Cn = Cn_n;
    }
}

// ---------------------------------------------------------------------------
// Launch. Workspace layout (bytes):
//   [0,16M)    xn f32          16 MB
//   [16M,24M)  xn bf16          8 MB
//   [24M,27M)  Wt bf16       2.06 MB
//   [27M,43M)  delta f32       16 MB
//   [43M,..)   bc f32        0.5 MB        total ~43.5 MB
// ---------------------------------------------------------------------------
extern "C" void kernel_launch(void* const* d_in, const int* in_sizes, int n_in,
                              void* d_out, int out_size, void* d_ws, size_t ws_size,
                              hipStream_t stream) {
    const float* x       = (const float*)d_in[0];
    const float* gamma   = (const float*)d_in[1];
    const float* beta    = (const float*)d_in[2];
    const float* W_delta = (const float*)d_in[3];
    const float* b_delta = (const float*)d_in[4];
    const float* W_B     = (const float*)d_in[5];
    const float* b_B     = (const float*)d_in[6];
    const float* W_C     = (const float*)d_in[7];
    const float* b_C     = (const float*)d_in[8];
    const float* A_log   = (const float*)d_in[9];
    const float* D_skip  = (const float*)d_in[10];
    float* out = (float*)d_out;

    char* ws = (char*)d_ws;
    float*          xn_f32 = (float*)(ws);
    unsigned short* xn_bf  = (unsigned short*)(ws + ((size_t)16 << 20));
    unsigned short* Wt     = (unsigned short*)(ws + ((size_t)24 << 20));
    float*          delta  = (float*)(ws + ((size_t)27 << 20));
    float*          bc     = (float*)(ws + ((size_t)43 << 20));

    pack_w_kernel<<<(NTOT * DDIM + 255) / 256, 256, 0, stream>>>(W_delta, W_B, W_C, Wt);
    ln_kernel<<<MROWS, 256, 0, stream>>>(x, gamma, beta, xn_f32, xn_bf);

    const int tiles32 = (MROWS / 32) * (NTOT / 32);  // 128*33 = 4224 waves
    gemm_kernel<<<tiles32 / 4, 128, 0, stream>>>(xn_bf, Wt, b_delta, b_B, b_C,
                                                 delta, bc);

    scan_kernel<<<(BSZ * DDIM * NST) / 256, 256, 0, stream>>>(
        x, xn_f32, delta, bc, A_log, D_skip, out);
}
